// ArchGNEncoder_29738353558365
// MI455X (gfx1250) — matrix operand order repaired and checked by reading, not compiled, not run on hardware
//
#include <hip/hip_runtime.h>

typedef __bf16 bf16;
typedef __attribute__((ext_vector_type(16))) __bf16 bf16x16;
typedef __attribute__((ext_vector_type(8)))  __bf16 bf16x8;
typedef __attribute__((ext_vector_type(8)))  float  f32x8;
typedef __attribute__((ext_vector_type(4)))  float  f32x4;

#define N_NODES 50000
#define HID     256
#define HEADS   4
#define HDIM    64
#define NTYPES  3
#define NEDGES  250000
#define NLAYERS 3

#define EPI_NONE       0
#define EPI_RESID_RELU 1

// Staged K-chunk depth and LDS A-tile geometry.
// Row = 64 bf16 (128B) padded to 72 (144B = 36 banks): 16-row x 16B fragment
// reads start at 16 distinct 4-bank groups covering all 64 banks -> conflict-free.
#define KCHUNK       64
#define LDS_STRIDE   72
#define LDS_ROWS_MAX 128
#define LDS_BUF_ELEMS (LDS_ROWS_MAX * LDS_STRIDE)

// ---------------------------------------------------------------------------
// Async global->LDS staging (GLOBAL_LOAD_ASYNC_TO_LDS_B128, ASYNCcnt-tracked)
// ---------------------------------------------------------------------------
__device__ __forceinline__ void async_b128(unsigned lds_off, unsigned long long g) {
    asm volatile("global_load_async_to_lds_b128 %0, %1, off"
                 :: "v"(lds_off), "v"(g)
                 : "memory");
}

// ---------------------------------------------------------------------------
// WMMA GEMM:  C[M,Nn] = epi(A_bf16[M,K] @ B + bias)
//   A : bf16 row-major, row stride lda, optional per-head offset (grid.z)
//   Bt: bf16 B^T [Nn][K] row-major (lane K-runs = contiguous 16B loads)
// Block = 256 threads = 8 waves arranged mWaves x nWaves (product == 8).
// Wave computes a 32x32 C macro-tile (2x2 of 16x16 WMMA tiles).
// A block-tile (mWaves*32 rows x 64 k) staged in LDS by async copies with a
// double buffer; per sync: 8 wmma + 8 ds_load_b128 + 8 global_load_b128 (B).
// Fragment layouts per CDNA5 ISA §7.12.2.
// ---------------------------------------------------------------------------
__global__ void wmma_gemm_bf16(
    const bf16* __restrict__ A, int lda, int aHeadOff,
    const bf16* __restrict__ Bt, int bHeadOff,
    const float* __restrict__ bias,
    float* __restrict__ Cf, bf16* __restrict__ Cb, int ldc, int cHeadOff,
    int M, int Nn, int K, int nWaves,
    int epi, const float* __restrict__ skipParam, const float* __restrict__ resid)
{
    __shared__ bf16 smA[2][LDS_BUF_ELEMS];

    const int tid  = threadIdx.x;
    const int wave = tid >> 5;
    const int lane = tid & 31;
    const int mWaves = 8 / nWaves;
    const int rows = mWaves * 32;              // A rows staged per block

    const int mBase = blockIdx.x * rows;
    const int n0 = (blockIdx.y * nWaves + (wave % nWaves)) * 32;
    const int hh = blockIdx.z;
    A  += (size_t)hh * aHeadOff;
    Bt += (size_t)hh * bHeadOff;

    const int ln   = lane & 15;
    const int half = lane >> 4;

    const bf16* bPtr0 = Bt + (size_t)(n0 + ln) * K;
    const bf16* bPtr1 = Bt + (size_t)(n0 + 16 + ln) * K;
    const int r0 = (wave / nWaves) * 32 + ln;  // local LDS row for m-tile 0

    // ---- loop-invariant async-copy assignments: nSegs == mWaves per thread ----
    const int nSegs = mWaves;                  // (rows*8 segs) / 256 threads
    unsigned long long gptr[4];
    unsigned loff[4];
#pragma unroll
    for (int j = 0; j < 4; ++j) {
        if (j < nSegs) {
            int s = tid + j * 256;
            int row = s >> 3, seg = s & 7;     // seg = 16B unit within 128B row-chunk
            int gr = mBase + row; if (gr >= M) gr = M - 1;
            gptr[j] = (unsigned long long)(size_t)(A + (size_t)gr * lda + seg * 8);
            loff[j] = (unsigned)(size_t)&smA[0][row * LDS_STRIDE + seg * 8];
        }
    }
    auto issueChunk = [&](int buf) {
#pragma unroll
        for (int j = 0; j < 4; ++j) {
            if (j < nSegs) {
                async_b128(loff[j] + (unsigned)buf * (LDS_BUF_ELEMS * 2), gptr[j]);
                gptr[j] += KCHUNK * 2;         // advance 128 bytes per chunk
            }
        }
    };

    f32x8 acc00 = {}, acc01 = {}, acc10 = {}, acc11 = {};

    issueChunk(0);
    const int nChunks = K / KCHUNK;
    for (int c = 0; c < nChunks; ++c) {
        const int cur = c & 1;
        const bool more = (c + 1 < nChunks);
        if (more) {
            issueChunk(cur ^ 1);
            // leave next chunk's copies in flight; ensure current chunk landed
            if (nSegs == 2) asm volatile("s_wait_asynccnt 0x2" ::: "memory");
            else            asm volatile("s_wait_asynccnt 0x4" ::: "memory");
        } else {
            asm volatile("s_wait_asynccnt 0x0" ::: "memory");
        }
        __syncthreads();

        const int k0 = c * KCHUNK;
#pragma unroll
        for (int kk = 0; kk < KCHUNK; kk += 32) {
            // A fragments from LDS (ds_load_b128, conflict-free padded stride)
            bf16x8 a00 = *(const bf16x8*)&smA[cur][ r0       * LDS_STRIDE + kk + half * 8];
            bf16x8 a01 = *(const bf16x8*)&smA[cur][ r0       * LDS_STRIDE + kk + 16 + half * 8];
            bf16x8 a10 = *(const bf16x8*)&smA[cur][(r0 + 16) * LDS_STRIDE + kk + half * 8];
            bf16x8 a11 = *(const bf16x8*)&smA[cur][(r0 + 16) * LDS_STRIDE + kk + 16 + half * 8];
            // B fragments direct from global (weights, L2/L1-resident)
            bf16x8 b00 = *(const bf16x8*)(bPtr0 + k0 + kk + half * 8);
            bf16x8 b01 = *(const bf16x8*)(bPtr0 + k0 + kk + 16 + half * 8);
            bf16x8 b10 = *(const bf16x8*)(bPtr1 + k0 + kk + half * 8);
            bf16x8 b11 = *(const bf16x8*)(bPtr1 + k0 + kk + 16 + half * 8);

            bf16x16 af0, af1, bf0, bf1;
#pragma unroll
            for (int i = 0; i < 8; ++i) {
                af0[i] = a00[i]; af0[i + 8] = a01[i];
                af1[i] = a10[i]; af1[i + 8] = a11[i];
                bf0[i] = b00[i]; bf0[i + 8] = b01[i];
                bf1[i] = b10[i]; bf1[i + 8] = b11[i];
            }
            acc00 = __builtin_amdgcn_wmma_f32_16x16x32_bf16(false, af0, false, bf0, (short)0, acc00, false, false);
            acc01 = __builtin_amdgcn_wmma_f32_16x16x32_bf16(false, af0, false, bf1, (short)0, acc01, false, false);
            acc10 = __builtin_amdgcn_wmma_f32_16x16x32_bf16(false, af1, false, bf0, (short)0, acc10, false, false);
            acc11 = __builtin_amdgcn_wmma_f32_16x16x32_bf16(false, af1, false, bf1, (short)0, acc11, false, false);
        }
        __syncthreads();   // protect cur buffer before it is overwritten
    }

    // ---- epilogue ----
    const int col0 = n0 + ln;
    const float bv0 = bias ? bias[col0] : 0.0f;
    const float bv1 = bias ? bias[col0 + 16] : 0.0f;
    float beta = 0.0f;
    if (epi == EPI_RESID_RELU) beta = 1.0f / (1.0f + __expf(-skipParam[0]));

    if (Cf) Cf += (size_t)hh * cHeadOff;
    if (Cb) Cb += (size_t)hh * cHeadOff;

    const int wm = wave / nWaves;
#pragma unroll
    for (int mi = 0; mi < 2; ++mi) {
        const f32x8 aN = mi ? acc10 : acc00;
        const f32x8 aP = mi ? acc11 : acc01;
#pragma unroll
        for (int i = 0; i < 8; ++i) {
            int r = mBase + wm * 32 + mi * 16 + half * 8 + i;
            if (r >= M) continue;
            size_t idx = (size_t)r * ldc + col0;
            float v0 = aN[i] + bv0;
            float v1 = aP[i] + bv1;
            if (epi == EPI_RESID_RELU) {
                v0 = beta * v0 + (1.0f - beta) * resid[idx];
                v1 = beta * v1 + (1.0f - beta) * resid[idx + 16];
                v0 = fmaxf(v0, 0.0f);
                v1 = fmaxf(v1, 0.0f);
            }
            if (Cf) { Cf[idx] = v0; Cf[idx + 16] = v1; }
            if (Cb) { Cb[idx] = (bf16)v0; Cb[idx + 16] = (bf16)v1; }
        }
    }
}

// ---------------------------------------------------------------------------
// Elementwise helpers
// ---------------------------------------------------------------------------
__global__ void k_zero_u32(unsigned* __restrict__ p, long n) {
    long i = (long)blockIdx.x * blockDim.x + threadIdx.x;
    if (i < n) p[i] = 0u;
}

__global__ void k_cvt_bf16(const float* __restrict__ in, bf16* __restrict__ out, long n) {
    long i = (long)blockIdx.x * blockDim.x + threadIdx.x;
    if (i < n) out[i] = (bf16)in[i];
}

__global__ void k_gelu_bf16(const float* __restrict__ in, bf16* __restrict__ out, long n) {
    long i = (long)blockIdx.x * blockDim.x + threadIdx.x;
    if (i < n) {
        float x = in[i];
        out[i] = (bf16)(0.5f * x * (1.0f + erff(x * 0.70710678118654752f)));
    }
}

// in: [batch][Kk][Nn] fp32  ->  out: [batch][Nn][Kk] bf16  (B^T for GEMM)
__global__ void k_transpose_bf16(const float* __restrict__ in, bf16* __restrict__ out,
                                 int Kk, int Nn, long total) {
    long i = (long)blockIdx.x * blockDim.x + threadIdx.x;
    if (i >= total) return;
    long mat = (long)Kk * Nn;
    long b = i / mat;
    long r = i - b * mat;
    int k = (int)(r / Nn), n = (int)(r % Nn);
    out[b * mat + (long)n * Kk + k] = (bf16)in[i];
}

// ---------------------------------------------------------------------------
// Edge phase
// ---------------------------------------------------------------------------
__device__ __forceinline__ unsigned fkey(float f) {
    unsigned u = __float_as_uint(f);
    return (u & 0x80000000u) ? ~u : (u | 0x80000000u);
}
__device__ __forceinline__ float funkey(unsigned k) {
    unsigned u = (k & 0x80000000u) ? (k & 0x7FFFFFFFu) : ~k;
    return __uint_as_float(u);
}

__global__ void k_edge_alpha(const int* __restrict__ ei, int E,
                             const float* __restrict__ q,
                             const float* __restrict__ krel,
                             const float* __restrict__ prel,
                             float* __restrict__ alpha,
                             unsigned* __restrict__ nmax)
{
    long i = (long)blockIdx.x * blockDim.x + threadIdx.x;
    if (i >= (long)E * HEADS) return;
    int e = (int)(i >> 2), hh = (int)(i & 3);
    int src = ei[e], dst = ei[E + e];
    const f32x4* qp = (const f32x4*)(q    + (size_t)dst * HID + hh * HDIM);
    const f32x4* kp = (const f32x4*)(krel + (size_t)src * HID + hh * HDIM);
    float s = 0.0f;
#pragma unroll
    for (int j = 0; j < HDIM / 4; ++j) {
        f32x4 a = qp[j], b = kp[j];
        s += a[0]*b[0] + a[1]*b[1] + a[2]*b[2] + a[3]*b[3];
    }
    s *= prel[hh] * 0.125f;  // p_rel * 1/sqrt(64)
    alpha[i] = s;
    atomicMax(&nmax[(size_t)dst * HEADS + hh], fkey(s));
}

__global__ void k_edge_norm(const int* __restrict__ ei, int E,
                            float* __restrict__ alpha,
                            const unsigned* __restrict__ nmax,
                            float* __restrict__ nsum)
{
    long i = (long)blockIdx.x * blockDim.x + threadIdx.x;
    if (i >= (long)E * HEADS) return;
    int e = (int)(i >> 2), hh = (int)(i & 3);
    int dst = ei[E + e];
    float m = funkey(nmax[(size_t)dst * HEADS + hh]);
    float ex = __expf(alpha[i] - m);
    alpha[i] = ex;
    atomicAdd(&nsum[(size_t)dst * HEADS + hh], ex);
}

__global__ void k_edge_scatter(const int* __restrict__ ei, int E,
                               const float* __restrict__ alpha,
                               const float* __restrict__ nsum,
                               const float* __restrict__ vrel,
                               float* __restrict__ out)
{
    long i = (long)blockIdx.x * blockDim.x + threadIdx.x;
    if (i >= (long)E * HEADS * 4) return;
    int qd = (int)(i & 3);
    int hh = (int)((i >> 2) & 3);
    int e  = (int)(i >> 4);
    int src = ei[e], dst = ei[E + e];
    float w = alpha[((long)e << 2) + hh] / (nsum[(size_t)dst * HEADS + hh] + 1e-16f);
    const f32x4* vp = (const f32x4*)(vrel + (size_t)src * HID + hh * HDIM + qd * 16);
    float* op = out + (size_t)dst * HID + hh * HDIM + qd * 16;
#pragma unroll
    for (int j = 0; j < 4; ++j) {
        f32x4 v = vp[j];
        atomicAdd(op + j * 4 + 0, v[0] * w);
        atomicAdd(op + j * 4 + 1, v[1] * w);
        atomicAdd(op + j * 4 + 2, v[2] * w);
        atomicAdd(op + j * 4 + 3, v[3] * w);
    }
}

// ---------------------------------------------------------------------------
// Host orchestration
// ---------------------------------------------------------------------------
extern "C" void kernel_launch(void* const* d_in, const int* in_sizes, int n_in,
                              void* d_out, int out_size, void* d_ws, size_t ws_size,
                              hipStream_t stream)
{
    (void)in_sizes; (void)n_in; (void)out_size; (void)ws_size;

    const float* x  = (const float*)d_in[0];
    const int* ei[NTYPES] = {(const int*)d_in[1], (const int*)d_in[2], (const int*)d_in[3]};
    const float* W_in = (const float*)d_in[4];
    const float* b_in = (const float*)d_in[5];
    // layer params: Wk,bk,Wq,bq,Wv,bv,Wa,ba,skip,a_rel,m_rel,p_rel
    auto LP = [&](int l, int j) -> const float* { return (const float*)d_in[6 + l * 12 + j]; };

    char* p = (char*)d_ws;
    auto alloc = [&](size_t bytes) -> char* {
        char* r = p; p += (bytes + 255) & ~(size_t)255; return r;
    };

    const long NH = (long)N_NODES * HID;
    float*    h      = (float*)alloc(NH * 4);
    float*    q      = (float*)alloc(NH * 4);
    float*    krel   = (float*)alloc(NH * 4);
    float*    vrel   = (float*)alloc(NH * 4);
    float*    outacc = (float*)alloc(NH * 4);
    bf16*     h_bf   = (bf16*)alloc(NH * 2);
    bf16*     k_bf   = (bf16*)alloc(NH * 2);
    bf16*     v_bf   = (bf16*)alloc(NH * 2);
    bf16*     g_bf   = (bf16*)alloc(NH * 2);     // x_bf at start, gelu(out) per layer
    float*    alpha  = (float*)alloc((long)NEDGES * HEADS * 4);
    unsigned* nmax   = (unsigned*)alloc((long)N_NODES * HEADS * 4);
    float*    nsum   = (float*)alloc((long)N_NODES * HEADS * 4);
    const long WMAT = (long)HID * HID;
    const long RMAT = (long)NTYPES * HEADS * HDIM * HDIM;
    bf16* WinT = (bf16*)alloc(WMAT * 2);
    bf16 *WkT[NLAYERS], *WqT[NLAYERS], *WvT[NLAYERS], *WaT[NLAYERS], *aT[NLAYERS], *mT[NLAYERS];
    for (int l = 0; l < NLAYERS; ++l) {
        WkT[l] = (bf16*)alloc(WMAT * 2);
        WqT[l] = (bf16*)alloc(WMAT * 2);
        WvT[l] = (bf16*)alloc(WMAT * 2);
        WaT[l] = (bf16*)alloc(WMAT * 2);
        aT[l]  = (bf16*)alloc(RMAT * 2);
        mT[l]  = (bf16*)alloc(RMAT * 2);
    }

    const int TB = 256;
    auto cdiv = [](long a, long b) -> int { return (int)((a + b - 1) / b); };

    // ---- conversions / weight transposes ----
    k_cvt_bf16<<<cdiv(NH, TB), TB, 0, stream>>>(x, g_bf, NH);
    k_transpose_bf16<<<cdiv(WMAT, TB), TB, 0, stream>>>(W_in, WinT, HID, HID, WMAT);
    for (int l = 0; l < NLAYERS; ++l) {
        k_transpose_bf16<<<cdiv(WMAT, TB), TB, 0, stream>>>(LP(l, 0), WkT[l], HID, HID, WMAT);
        k_transpose_bf16<<<cdiv(WMAT, TB), TB, 0, stream>>>(LP(l, 2), WqT[l], HID, HID, WMAT);
        k_transpose_bf16<<<cdiv(WMAT, TB), TB, 0, stream>>>(LP(l, 4), WvT[l], HID, HID, WMAT);
        k_transpose_bf16<<<cdiv(WMAT, TB), TB, 0, stream>>>(LP(l, 6), WaT[l], HID, HID, WMAT);
        k_transpose_bf16<<<cdiv(RMAT, TB), TB, 0, stream>>>(LP(l, 9),  aT[l], HDIM, HDIM, RMAT);
        k_transpose_bf16<<<cdiv(RMAT, TB), TB, 0, stream>>>(LP(l, 10), mT[l], HDIM, HDIM, RMAT);
    }

    // Big GEMM: 8 waves as 2x4 -> 64x128 block tile;  grid (ceil(M/64), 2)
    dim3 gridBig(cdiv(N_NODES, 64), HID / 128, 1);
    // Head GEMM: 8 waves as 4x2 -> 128x64 block tile; grid (ceil(M/128), 1, HEADS)
    dim3 gridHead(cdiv(N_NODES, 128), 1, HEADS);

    // ---- input projection: h = x @ W_in + b_in ----
    wmma_gemm_bf16<<<gridBig, 256, 0, stream>>>(
        g_bf, HID, 0, WinT, 0, b_in, h, h_bf, HID, 0,
        N_NODES, HID, HID, 4, EPI_NONE, nullptr, nullptr);

    // ---- layers ----
    for (int l = 0; l < NLAYERS; ++l) {
        wmma_gemm_bf16<<<gridBig, 256, 0, stream>>>(
            h_bf, HID, 0, WkT[l], 0, LP(l, 1), nullptr, k_bf, HID, 0,
            N_NODES, HID, HID, 4, EPI_NONE, nullptr, nullptr);
        wmma_gemm_bf16<<<gridBig, 256, 0, stream>>>(
            h_bf, HID, 0, WqT[l], 0, LP(l, 3), q, nullptr, HID, 0,
            N_NODES, HID, HID, 4, EPI_NONE, nullptr, nullptr);
        wmma_gemm_bf16<<<gridBig, 256, 0, stream>>>(
            h_bf, HID, 0, WvT[l], 0, LP(l, 5), nullptr, v_bf, HID, 0,
            N_NODES, HID, HID, 4, EPI_NONE, nullptr, nullptr);

        k_zero_u32<<<cdiv(NH, TB), TB, 0, stream>>>((unsigned*)outacc, NH);

        for (int t = 0; t < NTYPES; ++t) {
            wmma_gemm_bf16<<<gridHead, 256, 0, stream>>>(
                k_bf, HID, HDIM, aT[l] + (size_t)t * HEADS * HDIM * HDIM, HDIM * HDIM,
                nullptr, krel, nullptr, HID, HDIM,
                N_NODES, HDIM, HDIM, 2, EPI_NONE, nullptr, nullptr);
            wmma_gemm_bf16<<<gridHead, 256, 0, stream>>>(
                v_bf, HID, HDIM, mT[l] + (size_t)t * HEADS * HDIM * HDIM, HDIM * HDIM,
                nullptr, vrel, nullptr, HID, HDIM,
                N_NODES, HDIM, HDIM, 2, EPI_NONE, nullptr, nullptr);

            const long NHh = (long)N_NODES * HEADS;
            k_zero_u32<<<cdiv(NHh, TB), TB, 0, stream>>>(nmax, NHh);
            k_zero_u32<<<cdiv(NHh, TB), TB, 0, stream>>>((unsigned*)nsum, NHh);

            const long EH = (long)NEDGES * HEADS;
            k_edge_alpha<<<cdiv(EH, TB), TB, 0, stream>>>(
                ei[t], NEDGES, q, krel, LP(l, 11) + t * HEADS, alpha, nmax);
            k_edge_norm<<<cdiv(EH, TB), TB, 0, stream>>>(
                ei[t], NEDGES, alpha, nmax, nsum);
            k_edge_scatter<<<cdiv(EH * 4, TB), TB, 0, stream>>>(
                ei[t], NEDGES, alpha, nsum, vrel, outacc);
        }

        // o = gelu(out) @ Wa + ba ; h = relu(beta*o + (1-beta)*h)
        k_gelu_bf16<<<cdiv(NH, TB), TB, 0, stream>>>(outacc, g_bf, NH);
        float* cf = (l == NLAYERS - 1) ? (float*)d_out : h;
        wmma_gemm_bf16<<<gridBig, 256, 0, stream>>>(
            g_bf, HID, 0, WaT[l], 0, LP(l, 7), cf, h_bf, HID, 0,
            N_NODES, HID, HID, 4, EPI_RESID_RELU, LP(l, 8), h);
    }
}
